// SDPAModule_446676599325
// MI455X (gfx1250) — matrix operand order repaired and checked by reading, not compiled, not run on hardware
//
#include <hip/hip_runtime.h>
#include <hip/hip_bf16.h>

#define NUM_HEADS 32
#define NUM_KV    8
#define Q_MULT    4
#define HEAD_DIM  128
#define WINDOW    1024
#define SM_SCALE  0.08838834764831845f            // 1/sqrt(128)
#define LOG2E     1.4426950408889634f
#define QSCALE    (SM_SCALE * LOG2E)              // fold log2(e): v_exp_f32 is exp2

typedef __bf16 bf16;
typedef __bf16 v16bf __attribute__((ext_vector_type(16)));
typedef float  v8f   __attribute__((ext_vector_type(8)));

struct U32x8 { uint4 a, b; };
struct alignas(16) BH8  { bf16 h[8];  };
struct BH16 { bf16 h[16]; };

// Load a 16-element bf16 fragment from two 16B-aligned LDS locations.
__device__ __forceinline__ v16bf ld_frag(const bf16* p0, const bf16* p1) {
    U32x8 s;
    s.a = *(const uint4*)p0;
    s.b = *(const uint4*)p1;
    return __builtin_bit_cast(v16bf, s);
}

// Butterfly reductions confined to 16-lane halves of a wave32
// (xor masks < 16 never cross the half boundary).
__device__ __forceinline__ float redmax16(float v) {
#pragma unroll
    for (int m = 1; m < 16; m <<= 1) v = fmaxf(v, __shfl_xor(v, m, 32));
    return v;
}
__device__ __forceinline__ float redsum16(float v) {
#pragma unroll
    for (int m = 1; m < 16; m <<= 1) v += __shfl_xor(v, m, 32);
    return v;
}

__global__ __launch_bounds__(256)
void sdpa_swa_sink_kernel(const float* __restrict__ Q,
                          const float* __restrict__ K,
                          const float* __restrict__ V,
                          const float* __restrict__ sinks,
                          float* __restrict__ out,
                          int S) {
    // Double-buffered K (row-major [key][d], stride 136 breaks bank conflicts)
    __shared__ bf16 Kl[2][32][136];
    // Double-buffered V transposed [d][key], stride 40
    __shared__ bf16 Vt[2][HEAD_DIM][40];
    // Per-wave P tile [q][k] for the C-layout -> A-layout transpose
    __shared__ bf16 Pl[8][16][40];

    const int qt  = blockIdx.x;       // 32-query tile
    const int kv  = blockIdx.y;       // KV head
    const int bb  = blockIdx.z;       // batch
    const int q0  = qt * 32;

    const int tid  = threadIdx.x;
    const int wav  = tid >> 5;
    const int lane = tid & 31;
    const int half = lane >> 4;       // 16-lane half
    const int col  = lane & 15;

    const int mh   = wav & 3;         // q-head within KV group
    const int rt   = wav >> 2;        // 16-row sub-tile (0/1)
    const int qw   = q0 + rt * 16;    // first query row of this wave
    const int head = kv * Q_MULT + mh;

    // ---- Q fragments in WMMA A-layout, pre-scaled by (1/sqrt(d))*log2(e) ----
    v16bf aQ[4];
    {
        const float* qrow =
            Q + (((size_t)bb * S + (qw + col)) * NUM_HEADS + head) * HEAD_DIM;
#pragma unroll
        for (int s = 0; s < 4; ++s) {
            const int d0 = 32 * s + 16 * half;
            float fa[16];
            *(float4*)(fa + 0)  = *(const float4*)(qrow + d0);
            *(float4*)(fa + 4)  = *(const float4*)(qrow + d0 + 4);
            *(float4*)(fa + 8)  = *(const float4*)(qrow + d0 + 8);
            *(float4*)(fa + 12) = *(const float4*)(qrow + d0 + 12);
            BH16 t;
#pragma unroll
            for (int i = 0; i < 16; ++i) t.h[i] = (bf16)(fa[i] * QSCALE);
            aQ[s] = __builtin_bit_cast(v16bf, t);
        }
    }

    // ---- accumulators + online-softmax state (base-2 logit domain) ----
    v8f acc[8];
#pragma unroll
    for (int c = 0; c < 8; ++c)
#pragma unroll
        for (int r = 0; r < 8; ++r) acc[c][r] = 0.0f;

    const float sink2 = sinks[head] * LOG2E;
    float rmax[8], lsum[8];
#pragma unroll
    for (int r = 0; r < 8; ++r) { rmax[r] = sink2; lsum[r] = 1.0f; }

    int kstart = q0 - WINDOW; if (kstart < 0) kstart = 0;   // both multiples of 32
    const int kend = q0 + 32;
    const int nIt  = (kend - kstart) >> 5;

    // ---- staging thread mapping + stepped global pointers ----
    const int srow = tid >> 3;           // key row within tile
    const int sd0  = (tid & 7) * 16;     // 16 d-elements per thread
    const size_t stepF = 32 * (size_t)NUM_KV * HEAD_DIM;   // floats to next tile
    const float* kg =
        K + (((size_t)bb * S + (kstart + srow)) * NUM_KV + kv) * HEAD_DIM + sd0;
    const float* vg =
        V + (((size_t)bb * S + (kstart + srow)) * NUM_KV + kv) * HEAD_DIM + sd0;

    float kf[16], vf[16];

    // prologue: load + convert + store tile 0 into buffer 0
    {
        *(float4*)(kf + 0)  = *(const float4*)(kg + 0);
        *(float4*)(kf + 4)  = *(const float4*)(kg + 4);
        *(float4*)(kf + 8)  = *(const float4*)(kg + 8);
        *(float4*)(kf + 12) = *(const float4*)(kg + 12);
        *(float4*)(vf + 0)  = *(const float4*)(vg + 0);
        *(float4*)(vf + 4)  = *(const float4*)(vg + 4);
        *(float4*)(vf + 8)  = *(const float4*)(vg + 8);
        *(float4*)(vf + 12) = *(const float4*)(vg + 12);
        BH8 x0, x1;
#pragma unroll
        for (int i = 0; i < 8; ++i) { x0.h[i] = (bf16)kf[i]; x1.h[i] = (bf16)kf[8 + i]; }
        *(BH8*)&Kl[0][srow][sd0]     = x0;
        *(BH8*)&Kl[0][srow][sd0 + 8] = x1;
#pragma unroll
        for (int i = 0; i < 16; ++i) Vt[0][sd0 + i][srow] = (bf16)vf[i];
        kg += stepF; vg += stepF;
    }

    int buf = 0;
    for (int it = 0; it < nIt; ++it) {
        const int kb = kstart + (it << 5);
        __syncthreads();                      // staged buffer now visible

        // ---- issue next tile's global loads early (overlap with WMMA) ----
        const bool more = (it + 1 < nIt);
        if (more) {
            *(float4*)(kf + 0)  = *(const float4*)(kg + 0);
            *(float4*)(kf + 4)  = *(const float4*)(kg + 4);
            *(float4*)(kf + 8)  = *(const float4*)(kg + 8);
            *(float4*)(kf + 12) = *(const float4*)(kg + 12);
            *(float4*)(vf + 0)  = *(const float4*)(vg + 0);
            *(float4*)(vf + 4)  = *(const float4*)(vg + 4);
            *(float4*)(vf + 8)  = *(const float4*)(vg + 8);
            *(float4*)(vf + 12) = *(const float4*)(vg + 12);
            if (it + 2 < nIt) {               // global_prefetch_b8 for tile+2
                __builtin_prefetch(kg + stepF, 0, 0);
                __builtin_prefetch(vg + stepF, 0, 0);
            }
            kg += stepF; vg += stepF;
        }

        // ---- S = Q * K^T for two 16-key sub-tiles ----
        v8f sc0, sc1;
        {
            v8f s;
#pragma unroll
            for (int r = 0; r < 8; ++r) s[r] = 0.0f;
            v8f t = s;
#pragma unroll
            for (int ds = 0; ds < 4; ++ds) {
                const bf16* bp0 = &Kl[buf][col][32 * ds + 16 * half];
                v16bf bK0 = ld_frag(bp0, bp0 + 8);
                s = __builtin_amdgcn_wmma_f32_16x16x32_bf16(
                        false, aQ[ds], false, bK0, (short)0, s, false, false);
                const bf16* bp1 = &Kl[buf][16 + col][32 * ds + 16 * half];
                v16bf bK1 = ld_frag(bp1, bp1 + 8);
                t = __builtin_amdgcn_wmma_f32_16x16x32_bf16(
                        false, aQ[ds], false, bK1, (short)0, t, false, false);
            }
            sc0 = s; sc1 = t;
        }

        // ---- masking only on edge tiles (wave-uniform predicate) ----
        const bool need_mask = (kb + 31 > qw) || (kb <= qw + 15 - WINDOW);
        if (need_mask) {
#pragma unroll
            for (int r = 0; r < 8; ++r) {
                const int q  = qw + r + 8 * half;   // C-layout row
                const int k0 = kb + col;
                const int k1 = k0 + 16;
                if (!((k0 <= q) && (k0 > q - WINDOW))) sc0[r] = -__builtin_inff();
                if (!((k1 <= q) && (k1 > q - WINDOW))) sc1[r] = -__builtin_inff();
            }
        }

        // ---- online softmax (base-2) + P->LDS + rescale acc ----
#pragma unroll
        for (int r = 0; r < 8; ++r) {
            float tmax = redmax16(fmaxf(sc0[r], sc1[r]));
            float mnew = fmaxf(rmax[r], tmax);
            float fac  = __builtin_amdgcn_exp2f(rmax[r] - mnew);
            float p0   = __builtin_amdgcn_exp2f(sc0[r] - mnew);
            float p1   = __builtin_amdgcn_exp2f(sc1[r] - mnew);
            lsum[r] = lsum[r] * fac + redsum16(p0 + p1);
            rmax[r] = mnew;

            const int qr = r + 8 * half;
            Pl[wav][qr][col]      = (bf16)p0;
            Pl[wav][qr][col + 16] = (bf16)p1;
#pragma unroll
            for (int c = 0; c < 8; ++c) acc[c][r] *= fac;
        }

        // ---- O += P * V ----
        {
            const bf16* pp = &Pl[wav][col][16 * half];
            v16bf aP = ld_frag(pp, pp + 8);
#pragma unroll
            for (int c = 0; c < 8; ++c) {
                const bf16* vp = &Vt[buf][16 * c + col][16 * half];
                v16bf bV = ld_frag(vp, vp + 8);
                acc[c] = __builtin_amdgcn_wmma_f32_16x16x32_bf16(
                             false, aP, false, bV, (short)0, acc[c], false, false);
            }
        }

        // ---- convert + store next tile into the other buffer ----
        if (more) {
            const int nb = buf ^ 1;
            BH8 x0, x1;
#pragma unroll
            for (int i = 0; i < 8; ++i) { x0.h[i] = (bf16)kf[i]; x1.h[i] = (bf16)kf[8 + i]; }
            *(BH8*)&Kl[nb][srow][sd0]     = x0;
            *(BH8*)&Kl[nb][srow][sd0 + 8] = x1;
#pragma unroll
            for (int i = 0; i < 16; ++i) Vt[nb][sd0 + i][srow] = (bf16)vf[i];
        }
        buf ^= 1;
    }

    // ---- epilogue: normalize and store fp32 output ----
#pragma unroll
    for (int r = 0; r < 8; ++r) {
        const float inv = 1.0f / lsum[r];
        const int q = qw + r + 8 * half;
        float* orow =
            out + (((size_t)bb * S + q) * NUM_HEADS + head) * HEAD_DIM;
#pragma unroll
        for (int c = 0; c < 8; ++c) orow[16 * c + col] = acc[c][r] * inv;
    }
}

extern "C" void kernel_launch(void* const* d_in, const int* in_sizes, int n_in,
                              void* d_out, int out_size, void* d_ws, size_t ws_size,
                              hipStream_t stream) {
    (void)n_in; (void)out_size; (void)d_ws; (void)ws_size;
    const float* Q     = (const float*)d_in[0];
    const float* K     = (const float*)d_in[1];
    const float* V     = (const float*)d_in[2];
    const float* sinks = (const float*)d_in[3];
    float* out = (float*)d_out;

    const int S = 2048;
    const int B = in_sizes[0] / (S * NUM_HEADS * HEAD_DIM);

    dim3 grid(S / 32, NUM_KV, B);
    sdpa_swa_sink_kernel<<<grid, 256, 0, stream>>>(Q, K, V, sinks, out, S);
}
